// Decoder_66786741452955
// MI455X (gfx1250) — matrix operand order repaired and checked by reading, not compiled, not run on hardware
//
#include <hip/hip_runtime.h>
#include <hip/hip_fp16.h>

typedef __attribute__((ext_vector_type(16))) _Float16 v16h;
typedef __attribute__((ext_vector_type(8)))  _Float16 v8h;
typedef __attribute__((ext_vector_type(8)))  float    v8f;

#define BATCH   64
#define NHEADS  12
#define HEADC   64
#define DIMC    768
#define HWPOS   3136
#define HW4     784     // HWPOS / 4
#define LDIM    512
#define KTILES  24      // 768 / 32
#define LN_EPS  1e-5f

// -------------------------------------------------------------------------
// Kernel 1: single-query attention per (b, head).
// grid = 768 blocks (b*NHEADS+h), 256 threads (8 waves).
// All of x is streamed as per-lane float4 (global_load_b128, 512B/wave/instr)
// Phase 1: scores[pos] = sum_c q[c]*x[c,pos]  -> LDS (float4, coalesced)
// Phase 2: block softmax stats (max, sum of exp), weights kept in LDS
// Phase 3: cls[c] = (sum_pos w[pos]*x[c,pos]) / S   (8 channels per wave)
// -------------------------------------------------------------------------
__global__ __launch_bounds__(256)
void attn_pool_kernel(const float* __restrict__ x,
                      const float* __restrict__ q,
                      _Float16* __restrict__ clsh /* [64*768] */) {
    __shared__ float4 sw4[HW4];    // scores, then unnormalized weights
    __shared__ float  red[256];

    const int bh = blockIdx.x;
    const int b  = bh / NHEADS;
    const int h  = bh % NHEADS;
    const int t  = threadIdx.x;

    const float4* px4 = (const float4*)(x + (size_t)(b * DIMC + h * HEADC) * HWPOS);
    const float*  qh  = q + h * HEADC;

    // q is uniform per block -> compiler can scalarize these loads
    float qr[HEADC];
    #pragma unroll
    for (int c = 0; c < HEADC; ++c) qr[c] = qh[c];

    // ---- Phase 1: scores + local max ------------------------------------
    float lmax = -3.4e38f;
    for (int f = t; f < HW4; f += 256) {
        float4 s = make_float4(0.f, 0.f, 0.f, 0.f);
        #pragma unroll
        for (int c = 0; c < HEADC; ++c) {
            const float4 v = px4[(size_t)c * HW4 + f];
            s.x = fmaf(v.x, qr[c], s.x);
            s.y = fmaf(v.y, qr[c], s.y);
            s.z = fmaf(v.z, qr[c], s.z);
            s.w = fmaf(v.w, qr[c], s.w);
        }
        sw4[f] = s;
        lmax = fmaxf(lmax, fmaxf(fmaxf(s.x, s.y), fmaxf(s.z, s.w)));
    }
    red[t] = lmax;
    __syncthreads();
    #pragma unroll
    for (int off = 128; off > 0; off >>= 1) {
        if (t < off) red[t] = fmaxf(red[t], red[t + off]);
        __syncthreads();
    }
    const float M = red[0];
    __syncthreads();

    // ---- Phase 2: exp + sum ---------------------------------------------
    float lsum = 0.0f;
    for (int f = t; f < HW4; f += 256) {
        float4 s = sw4[f];
        s.x = __expf(s.x - M);
        s.y = __expf(s.y - M);
        s.z = __expf(s.z - M);
        s.w = __expf(s.w - M);
        sw4[f] = s;
        lsum += (s.x + s.y) + (s.z + s.w);
    }
    red[t] = lsum;
    __syncthreads();
    #pragma unroll
    for (int off = 128; off > 0; off >>= 1) {
        if (t < off) red[t] += red[t + off];
        __syncthreads();
    }
    const float invS = 1.0f / red[0];
    __syncthreads();

    // ---- Phase 3: weighted sum; wave w owns channels w*8 .. w*8+7 -------
    const int wv = t >> 5;          // wave index 0..7
    const int ln = t & 31;          // lane 0..31
    float acc[8];
    #pragma unroll
    for (int k = 0; k < 8; ++k) acc[k] = 0.0f;

    const float4* pxc4 = px4 + (size_t)(wv * 8) * HW4;
    for (int f = ln; f < HW4; f += 32) {
        const float4 e = sw4[f];
        #pragma unroll
        for (int k = 0; k < 8; ++k) {
            const float4 v = pxc4[(size_t)k * HW4 + f];
            acc[k] = fmaf(v.x, e.x, acc[k]);
            acc[k] = fmaf(v.y, e.y, acc[k]);
            acc[k] = fmaf(v.z, e.z, acc[k]);
            acc[k] = fmaf(v.w, e.w, acc[k]);
        }
    }
    #pragma unroll
    for (int k = 0; k < 8; ++k) {
        float v = acc[k];
        #pragma unroll
        for (int off = 16; off > 0; off >>= 1)
            v += __shfl_xor(v, off, 32);
        if (ln == 0)
            clsh[(size_t)b * DIMC + h * HEADC + wv * 8 + k] =
                (_Float16)(v * invS);
    }
}

// -------------------------------------------------------------------------
// Kernel 2: pack Wl (f32 [768 x 512] row-major) into f16 WMMA B-fragment
// order: packB[((tile_n*24 + kt)*32 + lane)*16 + i]
// where lane = (n&15) + 16*khalf, K within kt-block = khalf*16 + i.
// -------------------------------------------------------------------------
__global__ __launch_bounds__(256)
void pack_w_kernel(const float* __restrict__ Wl, _Float16* __restrict__ packB) {
    const int idx = blockIdx.x * 256 + threadIdx.x;
    if (idx >= DIMC * LDIM) return;
    const int k = idx / LDIM;
    const int n = idx % LDIM;
    const int tile_n = n >> 4, nl = n & 15;
    const int kt = k >> 5,  kr = k & 31;
    const int khalf = kr >> 4, i = kr & 15;
    const int lane = nl + (khalf << 4);
    packB[(((size_t)tile_n * KTILES + kt) * 32 + lane) * 16 + i] =
        (_Float16)Wl[idx];
}

// -------------------------------------------------------------------------
// Kernel 3: WMMA GEMM  y[64x512] = cls[64x768] * Wl[768x512] + bias
// One wave per 16x16 C tile: 4 M-tiles x 32 N-tiles = 128 waves.
// -------------------------------------------------------------------------
__global__ __launch_bounds__(32)
void gemm_wmma_kernel(const _Float16* __restrict__ clsh,
                      const _Float16* __restrict__ packB,
                      const float* __restrict__ bl,
                      float* __restrict__ y) {
    const int tile = blockIdx.x;        // 0..127
    const int tm = tile & 3;            // M tile (64/16)
    const int tn = tile >> 2;           // N tile (512/16)
    const int L  = threadIdx.x;         // lane 0..31
    const int rl = L & 15;
    const int khalf = L >> 4;

    const _Float16* arow  = clsh + (size_t)(tm * 16 + rl) * DIMC;
    const _Float16* bbase = packB + (size_t)tn * KTILES * 32 * 16;

    v8f c = {};
    for (int kt = 0; kt < KTILES; ++kt) {
        const int k0 = kt * 32;
        // A fragment: row rl; halves 0..15 -> K = k0+khalf*16+0..15
        v16h a = *(const v16h*)(arow + k0 + khalf * 16);
        // B fragment: pre-packed, 32 contiguous bytes per lane
        v16h bf = *(const v16h*)(bbase + ((size_t)kt * 32 + L) * 16);
        c = __builtin_amdgcn_wmma_f32_16x16x32_f16(
                false, a, false, bf, (short)0, c, false, false);
    }

    // C/D layout: VGPR j -> row tm*16 + j + 8*khalf, col tn*16 + (lane&15)
    const int col  = tn * 16 + rl;
    const int rbase = tm * 16 + (khalf ? 8 : 0);
    const float bias = bl[col];
    #pragma unroll
    for (int j = 0; j < 8; ++j)
        y[(size_t)(rbase + j) * LDIM + col] = c[j] + bias;
}

// -------------------------------------------------------------------------
// Kernel 4: LayerNorm over each row of y[64 x 512].
// -------------------------------------------------------------------------
__global__ __launch_bounds__(256)
void ln_kernel(const float* __restrict__ y,
               const float* __restrict__ gamma,
               const float* __restrict__ beta,
               float* __restrict__ out) {
    __shared__ float red[256];
    const int row = blockIdx.x;
    const int t   = threadIdx.x;

    const float v0 = y[(size_t)row * LDIM + t];
    const float v1 = y[(size_t)row * LDIM + t + 256];

    red[t] = v0 + v1;
    __syncthreads();
    #pragma unroll
    for (int off = 128; off > 0; off >>= 1) {
        if (t < off) red[t] += red[t + off];
        __syncthreads();
    }
    const float mu = red[0] * (1.0f / LDIM);
    __syncthreads();

    const float d0 = v0 - mu, d1 = v1 - mu;
    red[t] = d0 * d0 + d1 * d1;
    __syncthreads();
    #pragma unroll
    for (int off = 128; off > 0; off >>= 1) {
        if (t < off) red[t] += red[t + off];
        __syncthreads();
    }
    const float rstd = rsqrtf(red[0] * (1.0f / LDIM) + LN_EPS);

    out[(size_t)row * LDIM + t]       = d0 * rstd * gamma[t]       + beta[t];
    out[(size_t)row * LDIM + t + 256] = d1 * rstd * gamma[t + 256] + beta[t + 256];
}

// -------------------------------------------------------------------------
// Host-side launcher
// inputs (setup_inputs order): x, q, Wl, bl, gamma, beta
// -------------------------------------------------------------------------
extern "C" void kernel_launch(void* const* d_in, const int* in_sizes, int n_in,
                              void* d_out, int out_size, void* d_ws, size_t ws_size,
                              hipStream_t stream) {
    const float* x     = (const float*)d_in[0];
    const float* q     = (const float*)d_in[1];
    const float* Wl    = (const float*)d_in[2];
    const float* bl    = (const float*)d_in[3];
    const float* gamma = (const float*)d_in[4];
    const float* beta  = (const float*)d_in[5];
    float* out = (float*)d_out;

    // workspace layout (all 256B-aligned):
    //   clsh : 64*768 f16   =  98,304 B  @ 0
    //   packB: 768*512 f16  = 786,432 B  @ 98,304
    //   y    : 64*512 f32   = 131,072 B  @ 884,736
    char* ws = (char*)d_ws;
    _Float16* clsh  = (_Float16*)(ws);
    _Float16* packB = (_Float16*)(ws + 98304);
    float*    y     = (float*)(ws + 98304 + 786432);

    attn_pool_kernel<<<BATCH * NHEADS, 256, 0, stream>>>(x, q, clsh);
    pack_w_kernel<<<(DIMC * LDIM + 255) / 256, 256, 0, stream>>>(Wl, packB);
    gemm_wmma_kernel<<<128, 32, 0, stream>>>(clsh, packB, bl, y);
    ln_kernel<<<BATCH, 256, 0, stream>>>(y, gamma, beta, out);
}
